// LayoutLMv2Attention_6296422056366
// MI455X (gfx1250) — matrix operand order repaired and checked by reading, not compiled, not run on hardware
//
#include <hip/hip_runtime.h>
#include <hip/hip_bf16.h>

typedef __bf16 bf16_t;
typedef __attribute__((ext_vector_type(16))) __bf16 v16bf;
typedef __attribute__((ext_vector_type(8)))  float  v8f;

constexpr int kB  = 4;
constexpr int kS  = 1024;
constexpr int kH  = 768;
constexpr int kNH = 12;
constexpr int kHD = 64;
constexpr int kM  = kB * kS;   // 4096 rows total
constexpr float kEPS = 1e-12f;

#if defined(__gfx1250__)
#define USE_ASYNC_LDS 1
#else
#define USE_ASYNC_LDS 0
#endif

#if USE_ASYNC_LDS
// ASYNCcnt-tracked copy of 16 bytes/lane from global memory into LDS.
// VDST = per-lane LDS byte address, VADDR = 64-bit global address (ISA 10.x).
__device__ __forceinline__ void async_b128(void* lds, const void* gaddr) {
  asm volatile("global_load_async_to_lds_b128 %0, %1, off"
               :
               : "v"((unsigned)(uintptr_t)lds),
                 "v"((unsigned long long)(uintptr_t)gaddr)
               : "memory");
}
#define WAIT_ASYNC(n) asm volatile("s_wait_asynccnt %0" ::"n"(n) : "memory")
#else
#define WAIT_ASYNC(n) ((void)0)
#endif

__device__ __forceinline__ bf16_t f2bf(float f) {
  unsigned u = __builtin_bit_cast(unsigned, f);
  unsigned r = u + 0x7FFFu + ((u >> 16) & 1u);   // round-to-nearest-even
  unsigned short h = (unsigned short)(r >> 16);
  return __builtin_bit_cast(bf16_t, h);
}

__device__ __forceinline__ v8f wmma_bf16(const v16bf& a, const v16bf& b, const v8f& c) {
  return __builtin_amdgcn_wmma_f32_16x16x32_bf16(false, a, false, b, (short)0, c, false, false);
}

// Copy a 16x32 fp32 tile (rows strided by gstride) into LDS.
// Async path: 4 x global_load_async_to_lds_b128 (each lane moves 16B -> 2KB).
__device__ __forceinline__ void tile_copy_16x32(const float* __restrict__ g,
                                                size_t gstride, float* l, int lane) {
#if USE_ASYNC_LDS
#pragma unroll
  for (int i = 0; i < 4; ++i) {
    const int row = i * 4 + (lane >> 3);
    const int col = (lane & 7) * 4;
    async_b128(l + row * 32 + col, g + (size_t)row * gstride + col);
  }
#else
#pragma unroll
  for (int i = 0; i < 4; ++i) {
    const int row = i * 4 + (lane >> 3);
    const int col = (lane & 7) * 4;
    const float4 v = *(const float4*)(g + (size_t)row * gstride + col);
    *(float4*)(l + row * 32 + col) = v;
  }
#endif
}

// ---------------------------------------------------------------------------
// Kernel 1: fused QKV projection. X[4096,768] fp32 x W[768,768] -> bf16.
// z==0: Q (scaled by 1/sqrt(HD)) row-major [4096,768]
// z==1: K row-major [4096,768]
// z==2: V stored transposed as [b, h, hd, t]  (t contiguous) for PV WMMA.
// ---------------------------------------------------------------------------
__global__ void qkv_gemm_kernel(const float* __restrict__ X,
                                const float* __restrict__ Wq, const float* __restrict__ bq,
                                const float* __restrict__ Wk, const float* __restrict__ bk,
                                const float* __restrict__ Wv, const float* __restrict__ bv,
                                bf16_t* __restrict__ qout, bf16_t* __restrict__ kout,
                                bf16_t* __restrict__ vtout) {
  const int lane = threadIdx.x & 31;
  const int wave = threadIdx.x >> 5;
  const int hi   = lane >> 4;      // lane-half (K split)
  const int ln   = lane & 15;
  const int m0   = blockIdx.x * 16;
  const int n0   = blockIdx.y * 128 + wave * 16;
  const int z    = blockIdx.z;

  const float* W    = (z == 0) ? Wq : (z == 1) ? Wk : Wv;
  const float* bias = (z == 0) ? bq : (z == 1) ? bk : bv;

  v8f acc = {};
  const float* arow = X + (size_t)(m0 + ln) * kH;

  for (int k0 = 0; k0 < kH; k0 += 32) {
    v16bf a, b;
    // A 16x32 bf16: lane = row, elem e -> K = k0 + hi*16 + e
    const float* ap = arow + k0 + hi * 16;
#pragma unroll
    for (int e = 0; e < 8; ++e) a[e] = f2bf(ap[e]);
#pragma unroll
    for (int e = 0; e < 8; ++e) a[8 + e] = f2bf(ap[8 + e]);
    // B 32x16 bf16: lane = col n0+ln, elem e -> K = k0 + hi*16 + e (stride H in W)
    const float* bp = W + (size_t)(k0 + hi * 16) * kH + n0 + ln;
#pragma unroll
    for (int e = 0; e < 16; ++e) b[e] = f2bf(bp[(size_t)e * kH]);
    acc = wmma_bf16(a, b, acc);
  }

  const float scale = (z == 0) ? 0.125f : 1.0f;  // 1/sqrt(64)
#pragma unroll
  for (int r = 0; r < 8; ++r) {
    const int orow = m0 + r + hi * 8;   // C layout: VGPR r -> row r + 8*hi
    const int ocol = n0 + ln;
    const float v = (acc[r] + bias[ocol]) * scale;
    if (z == 0) {
      qout[(size_t)orow * kH + ocol] = f2bf(v);
    } else if (z == 1) {
      kout[(size_t)orow * kH + ocol] = f2bf(v);
    } else {
      const int bb = orow / kS, s = orow % kS;
      const int hh = ocol / kHD, hd = ocol % kHD;
      vtout[(((size_t)bb * kNH + hh) * kHD + hd) * kS + s] = f2bf(v);
    }
  }
}

// ---------------------------------------------------------------------------
// Kernel 2: flash attention. One wave per (b, h, 16 q-rows).
// The 402MB rel_pos/rel_2d_pos stream sets the HBM roofline (~17us), so it is
// decoupled from compute: double-buffered global_load_async_to_lds_b128
// prefetch of the next 16x32 bias tiles (ASYNCcnt / s_wait_asynccnt) overlaps
// with the score WMMAs + online softmax + PV WMMAs of the current tile.
// ---------------------------------------------------------------------------
__global__ void flash_attn_kernel(const bf16_t* __restrict__ qb,
                                  const bf16_t* __restrict__ kb,
                                  const bf16_t* __restrict__ vtb,
                                  const float* __restrict__ relp,
                                  const float* __restrict__ rel2,
                                  bf16_t* __restrict__ ctx) {
  __shared__ float bias_lds[2][2][16 * 32];  // [buf][tensor][row*32+col]
  __shared__ bf16_t pl[16 * 32];             // P tile staging (row-major 16x32)

  const int lane = threadIdx.x & 31;
  const int hi = lane >> 4, ln = lane & 15;
  const int s0 = blockIdx.x * 16;
  const int hh = blockIdx.y;
  const int bb = blockIdx.z;

  // Q A-fragments for K(=hd) chunks 0..31 and 32..63
  const bf16_t* qp = qb + (size_t)(bb * kS + s0 + ln) * kH + hh * kHD;
  v16bf qf[2];
#pragma unroll
  for (int c = 0; c < 2; ++c) {
#pragma unroll
    for (int e = 0; e < 8; ++e) qf[c][e] = qp[c * 32 + hi * 16 + e];
#pragma unroll
    for (int e = 0; e < 8; ++e) qf[c][8 + e] = qp[c * 32 + hi * 16 + 8 + e];
  }

  v8f acc[4] = {v8f{}, v8f{}, v8f{}, v8f{}};
  float m[8], l[8];
#pragma unroll
  for (int r = 0; r < 8; ++r) { m[r] = -1e30f; l[r] = 0.0f; }

  const float* rp = relp + (((size_t)bb * kNH + hh) * kS + s0) * kS;
  const float* r2 = rel2 + (((size_t)bb * kNH + hh) * kS + s0) * kS;

  // warm-up: prefetch tile 0 into buffer 0
  tile_copy_16x32(rp, kS, &bias_lds[0][0][0], lane);
  tile_copy_16x32(r2, kS, &bias_lds[0][1][0], lane);

  for (int t0 = 0; t0 < kS; t0 += 32) {
    const int buf = (t0 >> 5) & 1;
    // prefetch next tile (wraps on last iter; harmless) then wait for current
    const int tn = (t0 + 32) & (kS - 1);
    tile_copy_16x32(rp + tn, kS, &bias_lds[buf ^ 1][0][0], lane);
    tile_copy_16x32(r2 + tn, kS, &bias_lds[buf ^ 1][1][0], lane);
    WAIT_ASYNC(8);        // 8 just issued outstanding; current buffer complete
    __syncthreads();
    const float* lrp = &bias_lds[buf][0][0];
    const float* lr2 = &bias_lds[buf][1][0];

    float sc[2][8];
#pragma unroll
    for (int tt = 0; tt < 2; ++tt) {
      const int t = t0 + tt * 16;
      // K^T B-fragments: lane = key t+ln, elem e -> hd = c*32 + hi*16 + e
      const bf16_t* kp = kb + (size_t)(bb * kS + t + ln) * kH + hh * kHD + hi * 16;
      v16bf kf0, kf1;
#pragma unroll
      for (int e = 0; e < 16; ++e) kf0[e] = kp[e];
#pragma unroll
      for (int e = 0; e < 16; ++e) kf1[e] = kp[32 + e];
      v8f sf = {};
      sf = wmma_bf16(qf[0], kf0, sf);
      sf = wmma_bf16(qf[1], kf1, sf);
#pragma unroll
      for (int r = 0; r < 8; ++r) {
        const int idx = (r + hi * 8) * 32 + tt * 16 + ln;
        sc[tt][r] = sf[r] + lrp[idx] + lr2[idx];
      }
    }
    // online softmax over this 32-wide block of keys
#pragma unroll
    for (int r = 0; r < 8; ++r) {
      float v = fmaxf(sc[0][r], sc[1][r]);
      v = fmaxf(v, __shfl_xor(v, 1, 16));
      v = fmaxf(v, __shfl_xor(v, 2, 16));
      v = fmaxf(v, __shfl_xor(v, 4, 16));
      v = fmaxf(v, __shfl_xor(v, 8, 16));
      const float nm = fmaxf(m[r], v);
      const float rescale = __expf(m[r] - nm);
      m[r] = nm;
      l[r] *= rescale;
#pragma unroll
      for (int c = 0; c < 4; ++c) acc[c][r] *= rescale;
      const float p0 = __expf(sc[0][r] - nm);
      const float p1 = __expf(sc[1][r] - nm);
      float rs = p0 + p1;
      rs += __shfl_xor(rs, 1, 16);
      rs += __shfl_xor(rs, 2, 16);
      rs += __shfl_xor(rs, 4, 16);
      rs += __shfl_xor(rs, 8, 16);
      l[r] += rs;
      const int rrow = r + hi * 8;
      pl[rrow * 32 + ln]      = f2bf(p0);
      pl[rrow * 32 + 16 + ln] = f2bf(p1);
    }
    __syncthreads();
    // Re-read P as an A-fragment (16x32)
    const bf16_t* pp = &pl[ln * 32];
    v16bf pf;
#pragma unroll
    for (int e = 0; e < 8; ++e) pf[e] = pp[hi * 16 + e];
#pragma unroll
    for (int e = 0; e < 8; ++e) pf[8 + e] = pp[hi * 16 + 8 + e];
    __syncthreads();
    // V B-fragments from transposed V: lane = hd col, elem e -> t = t0 + hi*16 + e
    const bf16_t* vbase = vtb + ((size_t)bb * kNH + hh) * kHD * (size_t)kS + t0 + hi * 16;
#pragma unroll
    for (int c = 0; c < 4; ++c) {
      const bf16_t* vp = vbase + (size_t)(c * 16 + ln) * kS;
      v16bf vf;
#pragma unroll
      for (int e = 0; e < 16; ++e) vf[e] = vp[e];
      acc[c] = wmma_bf16(pf, vf, acc[c]);
    }
  }

  // normalize and write context, row-major [4096, 768] bf16
#pragma unroll
  for (int c = 0; c < 4; ++c) {
#pragma unroll
    for (int r = 0; r < 8; ++r) {
      const int orow = s0 + r + hi * 8;
      const int ocol = hh * kHD + c * 16 + ln;
      ctx[(size_t)(bb * kS + orow) * kH + ocol] = f2bf(acc[c][r] / l[r]);
    }
  }
}

// ---------------------------------------------------------------------------
// Kernel 3: output projection + bias + residual -> Y fp32 (pre-LayerNorm)
// ---------------------------------------------------------------------------
__global__ void out_gemm_kernel(const bf16_t* __restrict__ ctx,
                                const float* __restrict__ Wo,
                                const float* __restrict__ bo,
                                const float* __restrict__ X,
                                float* __restrict__ Y) {
  const int lane = threadIdx.x & 31;
  const int wave = threadIdx.x >> 5;
  const int hi = lane >> 4, ln = lane & 15;
  const int m0 = blockIdx.x * 16;
  const int n0 = blockIdx.y * 128 + wave * 16;

  v8f acc = {};
  const bf16_t* arow = ctx + (size_t)(m0 + ln) * kH;
  for (int k0 = 0; k0 < kH; k0 += 32) {
    v16bf a, b;
    const bf16_t* ap = arow + k0 + hi * 16;
#pragma unroll
    for (int e = 0; e < 8; ++e) a[e] = ap[e];
#pragma unroll
    for (int e = 0; e < 8; ++e) a[8 + e] = ap[8 + e];
    const float* bp = Wo + (size_t)(k0 + hi * 16) * kH + n0 + ln;
#pragma unroll
    for (int e = 0; e < 16; ++e) b[e] = f2bf(bp[(size_t)e * kH]);
    acc = wmma_bf16(a, b, acc);
  }
#pragma unroll
  for (int r = 0; r < 8; ++r) {
    const int orow = m0 + r + hi * 8;
    const int ocol = n0 + ln;
    Y[(size_t)orow * kH + ocol] = acc[r] + bo[ocol] + X[(size_t)orow * kH + ocol];
  }
}

// ---------------------------------------------------------------------------
// Kernel 4: row LayerNorm. One 256-thread block (8 waves) per row of 768.
// ---------------------------------------------------------------------------
__global__ void layernorm_kernel(const float* __restrict__ Y,
                                 const float* __restrict__ g,
                                 const float* __restrict__ be,
                                 float* __restrict__ out) {
  __shared__ float red[8];
  const int row = blockIdx.x;
  const int tid = threadIdx.x;
  const float* y = Y + (size_t)row * kH;

  float vals[3];
  float s = 0.0f;
#pragma unroll
  for (int i = 0; i < 3; ++i) { vals[i] = y[tid + i * 256]; s += vals[i]; }
#pragma unroll
  for (int o = 16; o > 0; o >>= 1) s += __shfl_xor(s, o, 32);
  if ((tid & 31) == 0) red[tid >> 5] = s;
  __syncthreads();
  float tot = 0.0f;
#pragma unroll
  for (int i = 0; i < 8; ++i) tot += red[i];
  const float mu = tot * (1.0f / kH);
  __syncthreads();

  float v = 0.0f;
#pragma unroll
  for (int i = 0; i < 3; ++i) { const float d = vals[i] - mu; v += d * d; }
#pragma unroll
  for (int o = 16; o > 0; o >>= 1) v += __shfl_xor(v, o, 32);
  if ((tid & 31) == 0) red[tid >> 5] = v;
  __syncthreads();
  float vtot = 0.0f;
#pragma unroll
  for (int i = 0; i < 8; ++i) vtot += red[i];
  const float rstd = rsqrtf(vtot * (1.0f / kH) + kEPS);

#pragma unroll
  for (int i = 0; i < 3; ++i) {
    const int c = tid + i * 256;
    out[(size_t)row * kH + c] = (vals[i] - mu) * rstd * g[c] + be[c];
  }
}

extern "C" void kernel_launch(void* const* d_in, const int* in_sizes, int n_in,
                              void* d_out, int out_size, void* d_ws, size_t ws_size,
                              hipStream_t stream) {
  const float* X    = (const float*)d_in[0];
  const float* relp = (const float*)d_in[1];
  const float* rel2 = (const float*)d_in[2];
  const float* Wq   = (const float*)d_in[3];
  const float* bq   = (const float*)d_in[4];
  const float* Wk   = (const float*)d_in[5];
  const float* bk   = (const float*)d_in[6];
  const float* Wv   = (const float*)d_in[7];
  const float* bv   = (const float*)d_in[8];
  const float* Wo   = (const float*)d_in[9];
  const float* bo   = (const float*)d_in[10];
  const float* lng  = (const float*)d_in[11];
  const float* lnb  = (const float*)d_in[12];

  char* ws = (char*)d_ws;
  const size_t sz_bf = (size_t)kM * kH * sizeof(bf16_t);  // 6.29 MB each
  bf16_t* qb  = (bf16_t*)(ws);
  bf16_t* kb  = (bf16_t*)(ws + sz_bf);
  bf16_t* vtb = (bf16_t*)(ws + 2 * sz_bf);
  bf16_t* ctx = (bf16_t*)(ws + 3 * sz_bf);
  float*  Y   = (float*)(ws + 4 * sz_bf);                 // 12.6 MB fp32

  qkv_gemm_kernel<<<dim3(kM / 16, kH / 128, 3), 256, 0, stream>>>(
      X, Wq, bq, Wk, bk, Wv, bv, qb, kb, vtb);
  flash_attn_kernel<<<dim3(kS / 16, kNH, kB), 32, 0, stream>>>(
      qb, kb, vtb, relp, rel2, ctx);
  out_gemm_kernel<<<dim3(kM / 16, kH / 128), 256, 0, stream>>>(ctx, Wo, bo, X, Y);
  layernorm_kernel<<<kM, 256, 0, stream>>>(Y, lng, lnb, (float*)d_out);
}
